// NNUEModel_67645734912397
// MI455X (gfx1250) — compile-verified, matrix operand order checked by reading, not run on hardware
//
#include <hip/hip_runtime.h>

typedef __attribute__((ext_vector_type(16))) _Float16 v16h;
typedef __attribute__((ext_vector_type(8)))  float    v8f;

#define L1D    3072
#define L2D    15
#define L3D    32
#define NSTACK 8
#define NPSQT  8
#define FT_OUT 3080   // L1D + NPSQT
#define NA     32
#define NGEMM  144    // 128 (l1_w rows) + 16 (l1f_w rows)
#define KTILES 96     // 3072 / 32
#define NTILES 9      // 144 / 16
#define KC     1024   // K-chunk staged in LDS (halves per row)
#define KCT    32     // k-tiles per chunk

// ---------------------------------------------------------------------------
// Kernel 1: pack [l1_w (128x3072); l1f_w (16x3072)] into WMMA B-fragment
// layout, f16. For tile (kt, nt): lane l holds column n = nt*16 + (l&15);
// halves i follow the documented 16-bit operand layout:
//   half = l>>4;  kk(i) = (i<8) ? i + 8*half : i + 8 + 8*half
// so each lane's 16 halves are one contiguous 32-byte store.
// ---------------------------------------------------------------------------
__global__ void pack_w_kernel(const float* __restrict__ l1_w,
                              const float* __restrict__ l1f_w,
                              _Float16* __restrict__ pW) {
    int tid = blockIdx.x * blockDim.x + threadIdx.x;
    int total = KTILES * NTILES * 32;
    if (tid >= total) return;
    int lane = tid & 31;
    int t    = tid >> 5;
    int nt   = t % NTILES;
    int kt   = t / NTILES;
    int n    = nt * 16 + (lane & 15);
    int half = (lane >> 4) & 1;
    const float* src = (n < 128) ? (l1_w  + (size_t)n * L1D)
                                 : (l1f_w + (size_t)(n - 128) * L1D);
    _Float16* dst = pW + ((size_t)(kt * NTILES + nt) * 32 + lane) * 16;
#pragma unroll
    for (int i = 0; i < 16; ++i) {
        int kk = (i < 8) ? (i + half * 8) : (i + 8 + half * 8);
        dst[i] = (_Float16)src[kt * 32 + kk];
    }
}

// ---------------------------------------------------------------------------
// Kernel 2: feature transform for both perspectives + l0 activation +
// pairwise product + psqt term. One block per batch element; thread t owns
// columns [12t, 12t+12) (3 x float4 per gathered row => coalesced).
// Bandwidth-dominant: ~3.2 GB of gathered rows per call; 277 MB table is
// mostly L2 (192 MB) resident across the 4096-element batch.
// ---------------------------------------------------------------------------
__global__ void __launch_bounds__(256)
ft_kernel(const float* __restrict__ us,   const float* __restrict__ them,
          const int*   __restrict__ wI,   const float* __restrict__ wV,
          const int*   __restrict__ bI,   const float* __restrict__ bV,
          const int*   __restrict__ pIdx,
          const float* __restrict__ ftw,  const float* __restrict__ ftb,
          _Float16* __restrict__ l0p16,   float* __restrict__ psqt_term) {
    int b = blockIdx.x;
    int t = threadIdx.x;

    __shared__ float l0w[L1D];           // us*w + them*b, clipped
    __shared__ float l0b[L1D];           // us*b + them*w, clipped
    __shared__ int   swi[NA]; __shared__ float swv[NA];
    __shared__ int   sbi[NA]; __shared__ float sbv[NA];
    __shared__ float wps[NPSQT], bps[NPSQT];

    if (t < NA) {
        swi[t] = wI[b * NA + t];  swv[t] = wV[b * NA + t];
        sbi[t] = bI[b * NA + t];  sbv[t] = bV[b * NA + t];
    }
    __syncthreads();

    const int colBase = t * 12;
    float accw[12], accb[12];
#pragma unroll
    for (int j = 0; j < 12; ++j) { accw[j] = 0.f; accb[j] = 0.f; }
    float accwp = 0.f, accbp = 0.f;

    for (int a = 0; a < NA; ++a) {
        int idx = swi[a]; float v = swv[a];
        const float4* r =
            reinterpret_cast<const float4*>(ftw + (size_t)idx * FT_OUT + colBase);
        if (a + 1 < NA)
            __builtin_prefetch(ftw + (size_t)swi[a + 1] * FT_OUT + colBase, 0, 0);
#pragma unroll
        for (int q = 0; q < 3; ++q) {
            float4 x = r[q];
            accw[q*4+0] = fmaf(v, x.x, accw[q*4+0]);
            accw[q*4+1] = fmaf(v, x.y, accw[q*4+1]);
            accw[q*4+2] = fmaf(v, x.z, accw[q*4+2]);
            accw[q*4+3] = fmaf(v, x.w, accw[q*4+3]);
        }
        if (t < NPSQT)
            accwp = fmaf(v, ftw[(size_t)idx * FT_OUT + L1D + t], accwp);
    }
    for (int a = 0; a < NA; ++a) {
        int idx = sbi[a]; float v = sbv[a];
        const float4* r =
            reinterpret_cast<const float4*>(ftw + (size_t)idx * FT_OUT + colBase);
        if (a + 1 < NA)
            __builtin_prefetch(ftw + (size_t)sbi[a + 1] * FT_OUT + colBase, 0, 0);
#pragma unroll
        for (int q = 0; q < 3; ++q) {
            float4 x = r[q];
            accb[q*4+0] = fmaf(v, x.x, accb[q*4+0]);
            accb[q*4+1] = fmaf(v, x.y, accb[q*4+1]);
            accb[q*4+2] = fmaf(v, x.z, accb[q*4+2]);
            accb[q*4+3] = fmaf(v, x.w, accb[q*4+3]);
        }
        if (t < NPSQT)
            accbp = fmaf(v, ftw[(size_t)idx * FT_OUT + L1D + t], accbp);
    }

    float u = us[b], th = them[b];
#pragma unroll
    for (int j = 0; j < 12; ++j) {
        int col = colBase + j;
        float bias = ftb[col];
        float wv = accw[j] + bias;
        float bv = accb[j] + bias;
        l0w[col] = fminf(fmaxf(u * wv + th * bv, 0.f), 1.f);
        l0b[col] = fminf(fmaxf(u * bv + th * wv, 0.f), 1.f);
    }
    if (t < NPSQT) {
        float bias = ftb[L1D + t];
        wps[t] = accwp + bias;
        bps[t] = accbp + bias;
    }
    __syncthreads();

    _Float16* outp = l0p16 + (size_t)b * L1D;
#pragma unroll
    for (int j = 0; j < 12; ++j) {
        int col = colBase + j;
        float p = (col < 1536) ? l0w[col] * l0w[col + 1536]
                               : l0b[col - 1536] * l0b[col];
        outp[col] = (_Float16)(p * (127.f / 128.f));
    }
    if (t == 0) {
        int p = pIdx[b];
        psqt_term[b] = (wps[p] - bps[p]) * (u - 0.5f);
    }
}

// ---------------------------------------------------------------------------
// Kernel 3: Y(B x 144) = l0p16(B x 3072) @ Wpacked, f16 inputs, f32 accum.
// One block = 16 batch rows, 9 waves; wave w owns N-tile w (16 cols).
// A slab (16 x 1024 halves) staged in LDS per K-chunk; 96 WMMAs per wave.
// Two independent accumulator chains (even/odd k-tile) so each iteration
// issues 4 global_load_b128 + 4 ds_load_b128, waits once, then runs two
// back-to-back independent v_wmma ops (better latency overlap per wave).
// ---------------------------------------------------------------------------
__global__ void __launch_bounds__(288)
gemm_l1_kernel(const _Float16* __restrict__ X,
               const _Float16* __restrict__ pW,
               float* __restrict__ Y) {
    __shared__ __align__(16) _Float16 As[16 * KC];   // 32 KB
    const int mbase = blockIdx.x * 16;
    const int tid   = threadIdx.x;
    const int wave  = tid >> 5;          // 0..8 -> N tile
    const int lane  = tid & 31;
    const int half  = lane >> 4;         // K-half selector
    const int rowL  = lane & 15;         // M row (A) / N col (B,C)

    v8f c0 = {}, c1 = {};

    for (int kc = 0; kc < L1D / KC; ++kc) {
        // cooperative global->LDS copy of 16 rows x KC halves (uint4 = 8 halves)
        for (int idx = tid; idx < 16 * (KC / 8); idx += 288) {
            int r = idx >> 7, q = idx & 127;
            reinterpret_cast<uint4*>(As)[idx] =
                reinterpret_cast<const uint4*>(
                    X + (size_t)(mbase + r) * L1D + kc * KC)[q];
        }
        __syncthreads();

#pragma unroll 1
        for (int ktl = 0; ktl < KCT; ktl += 2) {
            int koff0 = ktl * 32 + half * 8;
            int koff1 = koff0 + 32;
            union { uint4 u[2]; v16h h; } a0, a1;
            a0.u[0] = *reinterpret_cast<const uint4*>(&As[rowL * KC + koff0]);
            a0.u[1] = *reinterpret_cast<const uint4*>(&As[rowL * KC + koff0 + 16]);
            a1.u[0] = *reinterpret_cast<const uint4*>(&As[rowL * KC + koff1]);
            a1.u[1] = *reinterpret_cast<const uint4*>(&As[rowL * KC + koff1 + 16]);

            int ktg = kc * KCT + ktl;
            v16h b0 = *reinterpret_cast<const v16h*>(
                pW + ((size_t)((ktg    ) * NTILES + wave) * 32 + lane) * 16);
            v16h b1 = *reinterpret_cast<const v16h*>(
                pW + ((size_t)((ktg + 1) * NTILES + wave) * 32 + lane) * 16);

            c0 = __builtin_amdgcn_wmma_f32_16x16x32_f16(
                     false, a0.h, false, b0, (short)0, c0, false, false);
            c1 = __builtin_amdgcn_wmma_f32_16x16x32_f16(
                     false, a1.h, false, b1, (short)0, c1, false, false);
        }
        __syncthreads();
    }

    v8f c = c0 + c1;

    // C layout: VGPR r -> M = r + 8*half, N = lane&15
#pragma unroll
    for (int r = 0; r < 8; ++r) {
        int m = r + half * 8;
        Y[(size_t)(mbase + m) * NGEMM + wave * 16 + rowL] = c[r];
    }
}

// ---------------------------------------------------------------------------
// Kernel 4: tiny tail (stack select, L2=15 squared-clip concat, L3=32, out).
// One thread per batch element (~1k FMAs each).
// ---------------------------------------------------------------------------
__global__ void tail_kernel(const float* __restrict__ Y,
                            const int*   __restrict__ lsIdx,
                            const float* __restrict__ l1_b,
                            const float* __restrict__ l1f_b,
                            const float* __restrict__ l2_w,
                            const float* __restrict__ l2_b,
                            const float* __restrict__ out_w,
                            const float* __restrict__ out_b,
                            const float* __restrict__ psqt_term,
                            float* __restrict__ out, int Bdim) {
    int b = blockIdx.x * blockDim.x + threadIdx.x;
    if (b >= Bdim) return;
    int s = lsIdx[b];
    const float* yb = Y + (size_t)b * NGEMM;

    float l1x[2 * L2D];
#pragma unroll
    for (int j = 0; j < L2D; ++j) {
        float cj = yb[s * 16 + j] + l1_b[s * 16 + j];
        float fj = yb[128 + j] + l1f_b[j];
        float tsum = cj + fj;
        float sq = tsum * tsum * (127.f / 128.f);
        l1x[j]       = fminf(fmaxf(sq,   0.f), 1.f);
        l1x[L2D + j] = fminf(fmaxf(tsum, 0.f), 1.f);
    }
    float l1c_out = yb[s * 16 + L2D] + l1_b[s * 16 + L2D];
    float l1f_out = yb[128 + L2D] + l1f_b[L2D];

    float acc = out_b[s];
#pragma unroll 1
    for (int r = 0; r < L3D; ++r) {
        const float* wr = l2_w + (size_t)(s * L3D + r) * (2 * L2D);
        float v = l2_b[s * L3D + r];
#pragma unroll
        for (int j = 0; j < 2 * L2D; ++j) v = fmaf(wr[j], l1x[j], v);
        v = fminf(fmaxf(v, 0.f), 1.f);
        acc = fmaf(out_w[s * L3D + r], v, acc);
    }
    out[b] = acc + l1f_out + l1c_out + psqt_term[b];
}

// ---------------------------------------------------------------------------
extern "C" void kernel_launch(void* const* d_in, const int* in_sizes, int n_in,
                              void* d_out, int out_size, void* d_ws, size_t ws_size,
                              hipStream_t stream) {
    const float* us   = (const float*)d_in[0];
    const float* them = (const float*)d_in[1];
    const int*   wI   = (const int*)  d_in[2];
    const float* wV   = (const float*)d_in[3];
    const int*   bI   = (const int*)  d_in[4];
    const float* bV   = (const float*)d_in[5];
    const int*   pIdx = (const int*)  d_in[6];
    const int*   lsI  = (const int*)  d_in[7];
    const float* ftw  = (const float*)d_in[8];
    const float* ftb  = (const float*)d_in[9];
    const float* l1w  = (const float*)d_in[10];
    const float* l1b  = (const float*)d_in[11];
    const float* l1fw = (const float*)d_in[12];
    const float* l1fb = (const float*)d_in[13];
    const float* l2w  = (const float*)d_in[14];
    const float* l2b  = (const float*)d_in[15];
    const float* ow   = (const float*)d_in[16];
    const float* ob   = (const float*)d_in[17];

    const int B = in_sizes[0];   // 4096

    char* ws = (char*)d_ws;
    size_t off = 0;
    _Float16* l0p16 = (_Float16*)(ws + off); off += (size_t)B * L1D * sizeof(_Float16);
    float*    Ybuf  = (float*)   (ws + off); off += (size_t)B * NGEMM * sizeof(float);
    _Float16* pW    = (_Float16*)(ws + off); off += (size_t)KTILES * NTILES * 32 * 16 * sizeof(_Float16);
    float*    psqtT = (float*)   (ws + off); off += (size_t)B * sizeof(float);

    {
        int total = KTILES * NTILES * 32;
        pack_w_kernel<<<(total + 255) / 256, 256, 0, stream>>>(l1w, l1fw, pW);
    }
    ft_kernel<<<B, 256, 0, stream>>>(us, them, wI, wV, bI, bV, pIdx,
                                     ftw, ftb, l0p16, psqtT);
    gemm_l1_kernel<<<B / 16, 288, 0, stream>>>(l0p16, pW, Ybuf);
    tail_kernel<<<(B + 255) / 256, 256, 0, stream>>>(Ybuf, lsI, l1b, l1fb,
                                                     l2w, l2b, ow, ob, psqtT,
                                                     (float*)d_out, B);
}
